// SimpleRNN_62955630625141
// MI455X (gfx1250) — matrix operand order repaired and checked by reading, not compiled, not run on hardware
//
#include <hip/hip_runtime.h>
#include <hip/hip_bf16.h>

typedef __attribute__((ext_vector_type(16))) __bf16 v16bf;
typedef __attribute__((ext_vector_type(8)))  __bf16 v8bf;
typedef __attribute__((ext_vector_type(8)))  float  v8f;
typedef __attribute__((__vector_size__(4 * sizeof(int)))) int v4i;

#define RNN_H 1024
#define RNN_B 64
#define RNN_T 512
#define RNN_V 256
#define RNN_O 1024
#define FC_K  2048
#define NWG_DIR 16
#define SCAN_WGS 32
#define NCOL_WG 64   /* RNN_H / NWG_DIR */
#define SCAN_LDS ((RNN_B*RNN_H + NCOL_WG*RNN_H) * 2 + 256)

#define AS1 __attribute__((address_space(1)))
#define AS3 __attribute__((address_space(3)))

static __device__ __forceinline__ unsigned short f32_to_bf16(float f) {
  unsigned u; __builtin_memcpy(&u, &f, 4);
  u += 0x7FFFu + ((u >> 16) & 1u);              // round-to-nearest-even
  return (unsigned short)(u >> 16);
}

// Hardware tanh (CDNA5 TRANS op) instead of libm's exp-based expansion.
static __device__ __forceinline__ float tanh_fast(float x) {
#if __has_builtin(__builtin_amdgcn_tanhf)
  return __builtin_amdgcn_tanhf(x);
#else
  float r;
  asm volatile("v_tanh_f32 %0, %1\n\tv_nop\n\tv_nop" : "=v"(r) : "v"(x));
  return r;
#endif
}

// 128KB contiguous global -> LDS stage by 512 threads (16B x 16 per thread).
// Prefer the gfx1250 async-to-LDS DMA path (ASYNCcnt) when the builtin exists.
static __device__ __forceinline__ void stage128k(const uint4* __restrict__ gsrc,
                                                 uint4* __restrict__ ldst, int tid) {
#if __has_builtin(__builtin_amdgcn_global_load_async_to_lds_b128)
  const AS1 uint4* g = (const AS1 uint4*)gsrc;
  AS3 uint4* l = (AS3 uint4*)ldst;
  #pragma unroll
  for (int j = 0; j < 16; ++j)
    __builtin_amdgcn_global_load_async_to_lds_b128(
        (AS1 v4i*)(g + tid + j * 512), (AS3 v4i*)(l + tid + j * 512), 0, 0);
#if __has_builtin(__builtin_amdgcn_s_wait_asynccnt)
  __builtin_amdgcn_s_wait_asynccnt(0);
#else
  asm volatile("s_wait_asynccnt 0x0" ::: "memory");
#endif
#else
  #pragma unroll
  for (int j = 0; j < 16; ++j) ldst[tid + j * 512] = gsrc[tid + j * 512];
#endif
}

// A fragment (16x32 bf16, wave32): lane l<16 = row l, K{k0..k0+15} contiguously;
// lanes 16-31 = row l-16, K{k0+16..k0+31}.
static __device__ __forceinline__ v16bf load_a_frag(const unsigned short* base, int lane,
                                                    int stride, int k0) {
  const int r  = lane & 15;
  const int kh = (lane >> 4) * 16;
  const v8bf lo = *(const v8bf*)(base + (size_t)r * stride + k0 + kh);
  const v8bf hi = *(const v8bf*)(base + (size_t)r * stride + k0 + kh + 8);
  return __builtin_shufflevector(lo, hi, 0,1,2,3,4,5,6,7,8,9,10,11,12,13,14,15);
}

// B fragment (32x16 bf16): lane n<16 = column n holding K{k0..k0+15} contiguously,
// lanes 16-31 = column n-16 holding K{k0+16..k0+31}. base is row-major [n][k].
static __device__ __forceinline__ v16bf load_b_frag(const unsigned short* base, int lane,
                                                    int stride, int k0) {
  const int c  = lane & 15;
  const int kh = (lane >> 4) * 16;
  return *(const v16bf*)(base + (size_t)c * stride + k0 + kh);
}

// ---- prep: xwrow[v][h] = W_ih[h][v] + b_ih[h] + b_hh[h] -------------------
__global__ void k_prep_xwrow(const float* __restrict__ Wih, const float* __restrict__ bih,
                             const float* __restrict__ bhh, float* __restrict__ dst) {
  const int v = blockIdx.x;
  #pragma unroll
  for (int i = 0; i < 4; ++i) {
    const int h = threadIdx.x + i * 256;
    dst[v * RNN_H + h] = Wih[h * RNN_V + v] + bih[h] + bhh[h];
  }
}

__global__ void k_cvt_bf16(const float* __restrict__ s, unsigned short* __restrict__ d, int n) {
  int i = blockIdx.x * blockDim.x + threadIdx.x;
  const int stride = gridDim.x * blockDim.x;
  for (; i < n; i += stride) d[i] = f32_to_bf16(s[i]);
}

__global__ void k_zero(unsigned short* __restrict__ h, unsigned* __restrict__ bar) {
  const int i = blockIdx.x * blockDim.x + threadIdx.x;
  if (i < 2 * 2 * RNN_B * RNN_H) h[i] = 0;
  if (i < 32) bar[i] = 0u;
}

// ---- persistent bidirectional RNN scan ------------------------------------
__global__ void __launch_bounds__(512)
k_rnn_scan(const int* __restrict__ x, const float* __restrict__ xwrow,
           const unsigned short* __restrict__ whh, unsigned short* __restrict__ hbuf,
           unsigned short* __restrict__ hs, unsigned* __restrict__ bar) {
  extern __shared__ char smem[];
  unsigned short* Alds = (unsigned short*)smem;          // 64 x 1024 halves (h_prev)
  unsigned short* Blds = Alds + RNN_B * RNN_H;           // 64 x 1024 halves (W_hh slice)
  int* toks = (int*)(Blds + NCOL_WG * RNN_H);            // 64 tokens for this step

  const int dir  = blockIdx.x >> 4;
  const int slot = blockIdx.x & 15;
  const int n0w  = slot * NCOL_WG;

  const unsigned short* whh_d = whh + (size_t)dir * RNN_H * RNN_H;
  const float* xw_d = xwrow + (size_t)dir * RNN_V * RNN_H;
  unsigned short* hb = hbuf + (size_t)dir * 2 * RNN_B * RNN_H;

  const int tid  = threadIdx.x;
  const int lane = tid & 31;
  const int wave = tid >> 5;
  const int mt = wave & 3, nt = wave >> 2;
  const int m0 = mt * 16;

  // preload W_hh rows n0w..n0w+63 (contiguous 128KB) into LDS, once
  stage128k((const uint4*)(whh_d + (size_t)n0w * RNN_H), (uint4*)Blds, tid);

  unsigned* cnt = bar;
  unsigned* gen = bar + 16;

  const int col   = lane & 15;
  const int rbase = (lane >> 4) * 8;
  const int n = n0w + nt * 16 + col;
  const unsigned short* abase = Alds + (size_t)m0 * RNN_H;
  const unsigned short* bbase = Blds + (size_t)(nt * 16) * RNN_H;

  for (int t = 0; t < RNN_T; ++t) {
    const int tt  = dir ? (RNN_T - 1 - t) : t;
    const int cur = t & 1, nxt = cur ^ 1;

    if (tid < RNN_B) toks[tid] = x[tid * RNN_T + tt];
    stage128k((const uint4*)(hb + (size_t)cur * RNN_B * RNN_H), (uint4*)Alds, tid);
    __syncthreads();

    // hoist the per-row token->xwrow gathers so they overlap the GEMM
    float xw_v[8];
    #pragma unroll
    for (int r = 0; r < 8; ++r) {
      const int tok = toks[m0 + rbase + r];
      xw_v[r] = xw_d[tok * RNN_H + n];
    }

    // software-pipelined K loop: fetch k+32 fragments before issuing WMMA k
    v8f acc = {};
    v16bf a = load_a_frag(abase, lane, RNN_H, 0);
    v16bf b = load_b_frag(bbase, lane, RNN_H, 0);
    #pragma unroll 4
    for (int k0 = 0; k0 < RNN_H - 32; k0 += 32) {
      v16bf a2 = load_a_frag(abase, lane, RNN_H, k0 + 32);
      v16bf b2 = load_b_frag(bbase, lane, RNN_H, k0 + 32);
      acc = __builtin_amdgcn_wmma_f32_16x16x32_bf16(false, a, false, b, (short)0, acc,
                                                    false, false);
      a = a2; b = b2;
    }
    acc = __builtin_amdgcn_wmma_f32_16x16x32_bf16(false, a, false, b, (short)0, acc,
                                                  false, false);

    // epilogue: add input contribution, hardware tanh, emit bf16
    unsigned short* hout = hb + (size_t)nxt * RNN_B * RNN_H;
    #pragma unroll
    for (int r = 0; r < 8; ++r) {
      const int row = m0 + rbase + r;
      const unsigned short v = f32_to_bf16(tanh_fast(acc[r] + xw_v[r]));
      hout[(size_t)row * RNN_H + n] = v;
      hs[((size_t)row * RNN_T + tt) * FC_K + dir * RNN_H + n] = v;
    }

    __syncthreads();          // Alds/toks reuse
    __threadfence();          // make this thread's h stores device-visible
    __syncthreads();
    if (tid == 0) {           // device-wide step barrier across the 32 WGs
      unsigned a0 = atomicAdd(cnt, 1u);
      if (a0 == SCAN_WGS - 1) {
        *cnt = 0u;
        __threadfence();
        atomicAdd(gen, 1u);
      } else {
        while (__hip_atomic_load(gen, __ATOMIC_ACQUIRE, __HIP_MEMORY_SCOPE_AGENT) <
               (unsigned)(t + 1)) {
          __builtin_amdgcn_s_sleep(2);
        }
      }
    }
    __syncthreads();
  }
}

// ---- FC: out[32768,1024] = hs[32768,2048](bf16) @ Wfc^T(bf16) + b ----------
// Each wave computes a 16x32 tile: one A fragment feeds two WMMAs per k-chunk.
__global__ void __launch_bounds__(256)
k_fc(const unsigned short* __restrict__ hs, const unsigned short* __restrict__ wfc,
     const float* __restrict__ bfc, float* __restrict__ out) {
  const int tid = threadIdx.x;
  const int lane = tid & 31, wave = tid >> 5;
  const int bm = blockIdx.x & 1023;    // 1024 blocks of 32 rows
  const int bn = blockIdx.x >> 10;     // 8 blocks of 128 cols
  const int wm = wave & 1, wn = wave >> 1;
  const int m0 = (bm * 2 + wm) * 16;
  const int n0 = (bn * 8 + wn * 2) * 16;     // this wave: cols [n0, n0+32)

  const unsigned short* abase  = hs  + (size_t)m0 * FC_K;
  const unsigned short* bbase0 = wfc + (size_t)n0 * FC_K;
  const unsigned short* bbase1 = wfc + (size_t)(n0 + 16) * FC_K;

  v8f acc0 = {}, acc1 = {};
  v16bf a  = load_a_frag(abase,  lane, FC_K, 0);
  v16bf b0 = load_b_frag(bbase0, lane, FC_K, 0);
  v16bf b1 = load_b_frag(bbase1, lane, FC_K, 0);
  #pragma unroll 2
  for (int k0 = 0; k0 < FC_K - 32; k0 += 32) {
    __builtin_prefetch(abase + (size_t)(lane & 15) * FC_K + k0 + 512, 0, 1);
    v16bf a2  = load_a_frag(abase,  lane, FC_K, k0 + 32);
    v16bf b02 = load_b_frag(bbase0, lane, FC_K, k0 + 32);
    v16bf b12 = load_b_frag(bbase1, lane, FC_K, k0 + 32);
    acc0 = __builtin_amdgcn_wmma_f32_16x16x32_bf16(false, a, false, b0, (short)0, acc0,
                                                   false, false);
    acc1 = __builtin_amdgcn_wmma_f32_16x16x32_bf16(false, a, false, b1, (short)0, acc1,
                                                   false, false);
    a = a2; b0 = b02; b1 = b12;
  }
  acc0 = __builtin_amdgcn_wmma_f32_16x16x32_bf16(false, a, false, b0, (short)0, acc0,
                                                 false, false);
  acc1 = __builtin_amdgcn_wmma_f32_16x16x32_bf16(false, a, false, b1, (short)0, acc1,
                                                 false, false);

  const int col = lane & 15, rbase = (lane >> 4) * 8;
  const float bias0 = bfc[n0 + col];
  const float bias1 = bfc[n0 + 16 + col];
  #pragma unroll
  for (int r = 0; r < 8; ++r) {
    const size_t rowoff = (size_t)(m0 + rbase + r) * RNN_O;
    out[rowoff + n0 + col]      = acc0[r] + bias0;
    out[rowoff + n0 + 16 + col] = acc1[r] + bias1;
  }
}

extern "C" void kernel_launch(void* const* d_in, const int* in_sizes, int n_in,
                              void* d_out, int out_size, void* d_ws, size_t ws_size,
                              hipStream_t stream) {
  (void)in_sizes; (void)n_in; (void)out_size; (void)ws_size;
  const int*   x     = (const int*)d_in[0];
  const float* Wih_f = (const float*)d_in[1];
  const float* Whh_f = (const float*)d_in[2];
  const float* bih_f = (const float*)d_in[3];
  const float* bhh_f = (const float*)d_in[4];
  const float* Wih_b = (const float*)d_in[5];
  const float* Whh_b = (const float*)d_in[6];
  const float* bih_b = (const float*)d_in[7];
  const float* bhh_b = (const float*)d_in[8];
  const float* Wfc   = (const float*)d_in[9];
  const float* bfc   = (const float*)d_in[10];
  float* out = (float*)d_out;

  // workspace layout (bytes): [0,2M) xwrow f32x2dir | [2M,6M) W_hh bf16 x2 |
  // [6M,10M) W_fc bf16 | [10M,10.5M) h ping-pong x2dir | +bar | [11M,139M) hs bf16
  char* ws = (char*)d_ws;
  float*          xwrow  = (float*)ws;
  unsigned short* whh_bf = (unsigned short*)(ws + (2u << 20));
  unsigned short* wfc_bf = (unsigned short*)(ws + (6u << 20));
  unsigned short* hbuf   = (unsigned short*)(ws + (10u << 20));
  unsigned*       bar    = (unsigned*)(ws + (10u << 20) + (512u << 10));
  unsigned short* hs     = (unsigned short*)(ws + (11u << 20));

  (void)hipFuncSetAttribute(reinterpret_cast<const void*>(&k_rnn_scan),
                            hipFuncAttributeMaxDynamicSharedMemorySize, SCAN_LDS);

  k_prep_xwrow<<<RNN_V, 256, 0, stream>>>(Wih_f, bih_f, bhh_f, xwrow);
  k_prep_xwrow<<<RNN_V, 256, 0, stream>>>(Wih_b, bih_b, bhh_b, xwrow + RNN_V * RNN_H);
  k_cvt_bf16<<<1024, 256, 0, stream>>>(Whh_f, whh_bf, RNN_H * RNN_H);
  k_cvt_bf16<<<1024, 256, 0, stream>>>(Whh_b, whh_bf + RNN_H * RNN_H, RNN_H * RNN_H);
  k_cvt_bf16<<<2048, 256, 0, stream>>>(Wfc, wfc_bf, RNN_O * FC_K);
  k_zero<<<1024, 256, 0, stream>>>(hbuf, bar);

  k_rnn_scan<<<SCAN_WGS, 512, SCAN_LDS, stream>>>(x, xwrow, whh_bf, hbuf, hs, bar);
  k_fc<<<8192, 256, 0, stream>>>(hs, wfc_bf, bfc, out);
}
